// MeSH_GCN_55473797595784
// MI455X (gfx1250) — matrix operand order repaired and checked by reading, not compiled, not run on hardware
//
#include <hip/hip_runtime.h>
#include <hip/hip_bf16.h>

// ---------------------------------------------------------------------------
// MI455X / gfx1250: wave32, WMMA bf16 16x16x32, TDM tile + gather-mode DMA,
// double-buffered LDS staging, TENSORcnt sync.
// ---------------------------------------------------------------------------

typedef __attribute__((ext_vector_type(16))) __bf16 v16bf;
typedef __attribute__((ext_vector_type(8)))  float  v8f;
typedef __attribute__((ext_vector_type(4)))  unsigned int u32x4;
typedef __attribute__((ext_vector_type(8)))  int i32x8;
typedef __attribute__((ext_vector_type(4)))  int i32x4;

// LDS row stride in ushorts: 32 payload + 8 pad = 80B (16B aligned, bank-spread).
// Matches TDM pad_enable with pad_interval=3 (16 DWORDs) and pad_amount=3 (4 DWORDs).
#define TSTR 40

union FragBF { v16bf v; uint4 q[2]; };

__device__ __forceinline__ unsigned short f2bf(float f) {
    union { float f; unsigned u; } cv; cv.f = f;
    unsigned u = cv.u;
    unsigned r = u + 0x7FFFu + ((u >> 16) & 1u);   // round-to-nearest-even
    return (unsigned short)(r >> 16);
}

// A fragment: lane L holds row M=L%16, half h=L/16: K = {8h..8h+7, 16+8h..+7}
__device__ __forceinline__ v16bf frag_a(const unsigned short* base, int m, int h) {
    FragBF f;
    f.q[0] = *(const uint4*)(base + m * TSTR + 8 * h);
    f.q[1] = *(const uint4*)(base + m * TSTR + 16 + 8 * h);
    return f.v;
}
// B fragment: LDS is [N][K]; lane L col N=L%16, K = {8h..8h+7, 16+8h..+7}
__device__ __forceinline__ v16bf frag_b(const unsigned short* base, int n, int h) {
    FragBF f;
    const unsigned short* p = base + n * TSTR + 8 * h;
    f.q[0] = *(const uint4*)(p);
    f.q[1] = *(const uint4*)(p + 16);
    return f.v;
}

__device__ __forceinline__ v8f wmma_bf16(v16bf a, v16bf b, v8f c) {
    return __builtin_amdgcn_wmma_f32_16x16x32_bf16(false, a, false, b, (short)0, c,
                                                   false, false);
}

// Generic LDS pointer: low 32 bits are the LDS byte offset (aperture layout).
__device__ __forceinline__ unsigned lds_addr_of(const void* p) {
    return (unsigned)(unsigned long long)p;
}

__device__ __forceinline__ void tdm_issue(u32x4 g0, i32x8 g1, i32x4 g2, i32x4 g3) {
#if __has_include(<hip/amd_detail/amd_gfx1250_TDM.h>)
    i32x8 gz8 = { 0, 0, 0, 0, 0, 0, 0, 0 };
    __builtin_amdgcn_tensor_load_to_lds(g0, g1, g2, g3, gz8, 0);
#else
    __builtin_amdgcn_tensor_load_to_lds(g0, g1, g2, g3, 0);
#endif
}

// ---------------------------------------------------------------------------
// TDM: 2-D bf16 tile load Global->LDS, tile-relative OOB zero-fill, padded
// LDS rows (stride TSTR). Issue from ONE wave only (TDM ignores EXEC).
// ---------------------------------------------------------------------------
__device__ __forceinline__ void tdm_load_tile_bf16(
    unsigned lds_byte_addr, const void* gsrc,
    unsigned tensor_d0, unsigned tensor_d1,
    unsigned tile_d0, unsigned tile_d1,
    unsigned long long stride_d0)
{
    unsigned long long ga = (unsigned long long)gsrc;
    u32x4 g0;
    g0[0] = 1u;                                           // count=1
    g0[1] = lds_byte_addr;
    g0[2] = (unsigned)ga;
    g0[3] = (unsigned)((ga >> 32) & 0x1FFFFFFu) | (2u << 30);  // addr[56:32] | type=2

    // data_size=1 (2B) | pad_enable | pad_interval=3 (16 DW) | pad_amount=3 (4 DW)
    unsigned w0 = (1u << 16) | (1u << 20) | (3u << 22) | (3u << 25);
    unsigned w1 = (tensor_d0 & 0xFFFFu) << 16;
    unsigned w2 = (tensor_d0 >> 16) | ((tensor_d1 & 0xFFFFu) << 16);
    unsigned w3 = (tensor_d1 >> 16) | (tile_d0 << 16);
    unsigned w4 = tile_d1;
    unsigned w5 = (unsigned)(stride_d0 & 0xFFFFFFFFull);
    unsigned w6 = (unsigned)((stride_d0 >> 32) & 0xFFFFull);
    i32x8 g1 = { (int)w0, (int)w1, (int)w2, (int)w3, (int)w4, (int)w5, (int)w6, 0 };
    i32x4 gz = { 0, 0, 0, 0 };
    tdm_issue(g0, g1, gz, gz);
}

// ---------------------------------------------------------------------------
// TDM gather mode: 16 rows selected by 16-bit row indices (descriptor groups
// 2/3), each row = tile_d0 contiguous bf16 at gsrc + row_index*stride_d0.
// Embedding-lookup-as-DMA: the im2col gather runs on the TDM engine.
// ---------------------------------------------------------------------------
__device__ __forceinline__ void tdm_gather16_bf16(
    unsigned lds_byte_addr, const void* gsrc,
    unsigned tensor_d0,                   // valid columns (OOB -> zero)
    unsigned tile_d0,                     // columns per row (32)
    unsigned long long stride_d0,         // row stride in elements
    unsigned num_rows,                    // row-index OOB bound
    const unsigned* idx8)                 // 8 words = 16 packed u16 indices
{
    unsigned long long ga = (unsigned long long)gsrc;
    u32x4 g0;
    g0[0] = 1u | (1u << 31);              // count=1 | gather_mode (16-bit idx)
    g0[1] = lds_byte_addr;
    g0[2] = (unsigned)ga;
    g0[3] = (unsigned)((ga >> 32) & 0x1FFFFFFu) | (2u << 30);

    unsigned w0 = (1u << 16) | (1u << 20) | (3u << 22) | (3u << 25);
    unsigned w1 = (tensor_d0 & 0xFFFFu) << 16;
    unsigned w2 = (tensor_d0 >> 16) | ((num_rows & 0xFFFFu) << 16);
    unsigned w3 = (num_rows >> 16) | (tile_d0 << 16);
    unsigned w4 = 16u;                    // tile_dim1 = #valid indices
    unsigned w5 = (unsigned)(stride_d0 & 0xFFFFFFFFull);
    unsigned w6 = (unsigned)((stride_d0 >> 32) & 0xFFFFull);
    i32x8 g1 = { (int)w0, (int)w1, (int)w2, (int)w3, (int)w4, (int)w5, (int)w6, 0 };
    i32x4 g2 = { (int)idx8[0], (int)idx8[1], (int)idx8[2], (int)idx8[3] };
    i32x4 g3 = { (int)idx8[4], (int)idx8[5], (int)idx8[6], (int)idx8[7] };
    tdm_issue(g0, g1, g2, g3);
}

// ---------------------------------------------------------------------------
// f32 -> bf16 converters (one-time passes; pure bandwidth)
// ---------------------------------------------------------------------------
__global__ void convert_kernel(const float* __restrict__ src,
                               unsigned short* __restrict__ dst, int n) {
    for (int i = blockIdx.x * blockDim.x + threadIdx.x; i < n;
         i += gridDim.x * blockDim.x)
        dst[i] = f2bf(src[i]);
}

__global__ void transpose_convert_kernel(const float* __restrict__ src, // K x N
                                         unsigned short* __restrict__ dst, // N x K
                                         int K, int N) {
    int total = K * N;
    for (int i = blockIdx.x * blockDim.x + threadIdx.x; i < total;
         i += gridDim.x * blockDim.x) {
        int k = i / N, n = i - k * N;
        dst[(size_t)n * K + k] = f2bf(src[i]);
    }
}

// Conv weights (256,1,kw,200) -> bf16 [256][kw*224], 24 zero-pad cols per row
// so every 32-wide K step of the conv GEMM stays inside one embedding row.
__global__ void pad_convert_w_kernel(const float* __restrict__ src,
                                     unsigned short* __restrict__ dst, int kw) {
    const int Kp = kw * 224;
    const int total = 256 * Kp;
    for (int i = blockIdx.x * blockDim.x + threadIdx.x; i < total;
         i += gridDim.x * blockDim.x) {
        int n  = i / Kp;
        int rp = i - n * Kp;
        int r  = rp / 224;
        int c  = rp - r * 224;
        dst[i] = (c < 200) ? f2bf(src[(n * kw + r) * 200 + c]) : (unsigned short)0;
    }
}

// ---------------------------------------------------------------------------
// TextCNN conv(k=3,4,5) as WMMA GEMM over padded K' = kw*224.
// A tile: TDM gather (16 tokens x 32 cols); B tile: TDM tile from padded
// weights. Both double-buffered; inner loop = descriptors + 2 WMMA only.
// grid = (128 time-tiles, 32 batch, 3 widths), block = 256 (8 waves).
// ---------------------------------------------------------------------------
__global__ __launch_bounds__(256) void conv_maxpool_kernel(
    const int* __restrict__ seq, const unsigned short* __restrict__ embbf,
    const unsigned short* __restrict__ w3p, const float* __restrict__ b3,
    const unsigned short* __restrict__ w4p, const float* __restrict__ b4,
    const unsigned short* __restrict__ w5p, const float* __restrict__ b5,
    float* __restrict__ xfeat)
{
    const int kw = 3 + blockIdx.z;
    const unsigned short* Wp = (kw == 3) ? w3p : (kw == 4) ? w4p : w5p;
    const float* Bv = (kw == 3) ? b3 : (kw == 4) ? b4 : b5;
    const int outoff = (kw - 3) * 256;
    const int b  = blockIdx.y;
    const int t0 = blockIdx.x * 16;
    const int Kp = kw * 224;
    const int Ksteps = kw * 7;

    __shared__ unsigned short As[2][16 * TSTR];
    __shared__ unsigned short Bs[2][256 * TSTR];
    __shared__ int toks[20];

    const int tid  = threadIdx.x;
    const int lane = tid & 31, wv = tid >> 5;
    const int h = lane >> 4, l16 = lane & 15;

    if (tid < 20) {
        int t = t0 + tid;
        // clamp: rows with t>=2048 only feed outputs the epilogue skips
        toks[tid] = seq[b * 2048 + (t < 2048 ? t : 2047)];
    }
    __syncthreads();

    if (tid < 32) {                                   // prologue: tiles for kk=0
        unsigned idx[8];
        #pragma unroll
        for (int j = 0; j < 8; ++j)
            idx[j] = ((unsigned)toks[2 * j] & 0xFFFFu) |
                     (((unsigned)toks[2 * j + 1] & 0xFFFFu) << 16);
        tdm_gather16_bf16(lds_addr_of(As[0]), embbf, 200, 32, 200, 50000, idx);
        tdm_load_tile_bf16(lds_addr_of(Bs[0]), Wp, Kp, 256, 32, 256, Kp);
    }

    v8f acc0 = {}; v8f acc1 = {};

    for (int kk = 0; kk < Ksteps; ++kk) {
        const int cur = kk & 1, nxt = cur ^ 1;
        if (tid < 32) {                               // wave-uniform issue + wait
            if (kk + 1 < Ksteps) {
                int kb = (kk + 1) << 5;
                int r  = kb / 224;                    // embedding row in window
                int c0 = kb - r * 224;                // column offset (<= 192)
                unsigned idx[8];
                #pragma unroll
                for (int j = 0; j < 8; ++j)
                    idx[j] = ((unsigned)toks[r + 2 * j] & 0xFFFFu) |
                             (((unsigned)toks[r + 2 * j + 1] & 0xFFFFu) << 16);
                tdm_gather16_bf16(lds_addr_of(As[nxt]), embbf + c0,
                                  (unsigned)(200 - c0), 32, 200, 50000, idx);
                tdm_load_tile_bf16(lds_addr_of(Bs[nxt]), Wp + kb,
                                   (unsigned)(Kp - kb), 256, 32, 256, Kp);
                __builtin_amdgcn_s_wait_tensorcnt((short)2);  // older pair done
            } else {
                __builtin_amdgcn_s_wait_tensorcnt((short)0);
            }
        }
        __syncthreads();                              // publish DMA'd tiles
        v16bf a   = frag_a(As[cur], l16, h);
        v16bf bb0 = frag_b(Bs[cur], wv * 32 + l16, h);
        v16bf bb1 = frag_b(Bs[cur], wv * 32 + 16 + l16, h);
        acc0 = wmma_bf16(a, bb0, acc0);
        acc1 = wmma_bf16(a, bb1, acc1);
        __syncthreads();                              // tiles free for reuse
    }

    #pragma unroll
    for (int tile = 0; tile < 2; ++tile) {
        int ch = wv * 32 + tile * 16 + l16;
        float bias = Bv[ch];
        v8f acc = tile ? acc1 : acc0;
        #pragma unroll
        for (int r = 0; r < 8; ++r) {
            int t = t0 + r + 8 * h;
            if (t + kw <= 2048) {
                float v = acc[r] + bias;
                v = v > 0.f ? v : 0.f;                // >=0: int-bit atomicMax exact
                atomicMax((int*)&xfeat[b * 768 + outoff + ch], __float_as_int(v));
            }
        }
    }
}

// ---------------------------------------------------------------------------
// WMMA GEMM, TDM double-buffered: C[row_off+m,n] = act(A@Wt^T + bias)
// A: M x K bf16; Wt: N x K bf16 (pre-transposed). tile 32M x 64N, K step 32.
// ---------------------------------------------------------------------------
__global__ __launch_bounds__(256) void gemm_tdm_kernel(
    const unsigned short* __restrict__ Abf, const unsigned short* __restrict__ Wt,
    const float* __restrict__ bias, float* __restrict__ C,
    int M, int K, int N, int ldc, int row_off, int do_relu)
{
    const int m0 = blockIdx.x * 32;
    const int n0 = blockIdx.y * 64;
    const int tid  = threadIdx.x;
    const int lane = tid & 31, wv = tid >> 5;
    const int ms = wv >> 2, ns = wv & 3;
    const int h = lane >> 4, l16 = lane & 15;

    __shared__ unsigned short As[2][32 * TSTR];
    __shared__ unsigned short Bs[2][64 * TSTR];

    const int Ksteps = K >> 5;
    v8f acc = {};

    if (tid < 32) {
        tdm_load_tile_bf16(lds_addr_of(As[0]), Abf + (size_t)m0 * K,
                           (unsigned)K, (unsigned)(M - m0), 32, 32, K);
        tdm_load_tile_bf16(lds_addr_of(Bs[0]), Wt + (size_t)n0 * K,
                           (unsigned)K, (unsigned)(N - n0), 32, 64, K);
    }

    for (int kk = 0; kk < Ksteps; ++kk) {
        const int cur = kk & 1, nxt = cur ^ 1;
        if (tid < 32) {
            if (kk + 1 < Ksteps) {
                int kb = (kk + 1) << 5;
                tdm_load_tile_bf16(lds_addr_of(As[nxt]), Abf + (size_t)m0 * K + kb,
                                   (unsigned)(K - kb), (unsigned)(M - m0), 32, 32, K);
                tdm_load_tile_bf16(lds_addr_of(Bs[nxt]), Wt + (size_t)n0 * K + kb,
                                   (unsigned)(K - kb), (unsigned)(N - n0), 32, 64, K);
                __builtin_amdgcn_s_wait_tensorcnt((short)2);
            } else {
                __builtin_amdgcn_s_wait_tensorcnt((short)0);
            }
        }
        __syncthreads();
        v16bf a = frag_a(&As[cur][0], ms * 16 + l16, h);
        v16bf b = frag_b(&Bs[cur][0], ns * 16 + l16, h);
        acc = wmma_bf16(a, b, acc);
        __syncthreads();
    }

    int gn = n0 + ns * 16 + l16;
    float bv = bias[gn];
    #pragma unroll
    for (int r = 0; r < 8; ++r) {
        int gm = m0 + ms * 16 + r + 8 * h;
        if (gm < M) {
            float v = acc[r] + bv;
            if (do_relu) v = v > 0.f ? v : 0.f;
            C[(size_t)(row_off + gm) * ldc + gn] = v;
        }
    }
}

// ---------------------------------------------------------------------------
// GCN aggregation: acc[dst[e], :] += feat[src[e], :]  (L2-resident f32 atomics)
// ---------------------------------------------------------------------------
__global__ void scatter_add_kernel(const int* __restrict__ src,
                                   const int* __restrict__ dst,
                                   const float* __restrict__ feat,
                                   float* __restrict__ acc,
                                   int E, int F, int l2f4)
{
    const int total = E << l2f4;
    const int mask = (1 << l2f4) - 1;
    for (int i = blockIdx.x * blockDim.x + threadIdx.x; i < total;
         i += gridDim.x * blockDim.x) {
        int e  = i >> l2f4;
        int f4 = (i & mask) << 2;
        int s = src[e], d = dst[e];
        const float4 v = *(const float4*)(feat + (size_t)s * F + f4);
        float* p = acc + (size_t)d * F + f4;
        atomicAdd(p + 0, v.x); atomicAdd(p + 1, v.y);
        atomicAdd(p + 2, v.z); atomicAdd(p + 3, v.w);
    }
}

__global__ void zero_kernel(float* __restrict__ p, int n) {
    for (int i = blockIdx.x * blockDim.x + threadIdx.x; i < n;
         i += gridDim.x * blockDim.x)
        p[i] = 0.f;
}

// ---------------------------------------------------------------------------
// In-place log_softmax over 768 columns; one 256-thread block per row.
// ---------------------------------------------------------------------------
__global__ __launch_bounds__(256) void logsoftmax_kernel(float* __restrict__ out) {
    const int cols = 768;
    float* p = out + (size_t)blockIdx.x * cols;
    const int tid = threadIdx.x;

    float x[3];
    float m = -3.402823466e38f;
    #pragma unroll
    for (int j = 0; j < 3; ++j) { x[j] = p[tid + j * 256]; m = fmaxf(m, x[j]); }

    __shared__ float red[256];
    red[tid] = m; __syncthreads();
    #pragma unroll
    for (int s = 128; s > 0; s >>= 1) {
        if (tid < s) red[tid] = fmaxf(red[tid], red[tid + s]);
        __syncthreads();
    }
    m = red[0]; __syncthreads();

    float sum = 0.f;
    #pragma unroll
    for (int j = 0; j < 3; ++j) sum += __expf(x[j] - m);
    red[tid] = sum; __syncthreads();
    #pragma unroll
    for (int s = 128; s > 0; s >>= 1) {
        if (tid < s) red[tid] += red[tid + s];
        __syncthreads();
    }
    float ls = __logf(red[0]);
    #pragma unroll
    for (int j = 0; j < 3; ++j) p[tid + j * 256] = x[j] - m - ls;
}

// ---------------------------------------------------------------------------
extern "C" void kernel_launch(void* const* d_in, const int* in_sizes, int n_in,
                              void* d_out, int out_size, void* d_ws, size_t ws_size,
                              hipStream_t stream) {
    const int*   input_seq = (const int*)  d_in[0];   // (32, 2048)
    const int*   edge_src  = (const int*)  d_in[1];   // (600000,)
    const int*   edge_dst  = (const int*)  d_in[2];   // (600000,)
    const float* features  = (const float*)d_in[3];   // (50000, 128)
    const float* emb       = (const float*)d_in[4];   // (50000, 200)
    const float* w3 = (const float*)d_in[5];  const float* b3 = (const float*)d_in[6];
    const float* w4 = (const float*)d_in[7];  const float* b4 = (const float*)d_in[8];
    const float* w5 = (const float*)d_in[9];  const float* b5 = (const float*)d_in[10];
    const float* g1w = (const float*)d_in[11]; const float* g1b = (const float*)d_in[12];
    const float* g2w = (const float*)d_in[13]; const float* g2b = (const float*)d_in[14];
    float* out = (float*)d_out;                       // (50032, 768)

    const int NN = 50000, E = 600000;

    char* ws = (char*)d_ws;
    float* agg1 = (float*)ws;            ws += (size_t)NN * 128 * 4;
    float* hbuf = (float*)ws;            ws += (size_t)NN * 256 * 4;
    float* agg2 = (float*)ws;            ws += (size_t)NN * 256 * 4;
    unsigned short* embbf = (unsigned short*)ws; ws += (size_t)NN * 200 * 2;
    unsigned short* abf1  = (unsigned short*)ws; ws += (size_t)NN * 128 * 2;
    unsigned short* abf2  = (unsigned short*)ws; ws += (size_t)NN * 256 * 2;
    unsigned short* w3p   = (unsigned short*)ws; ws += 256 * (3 * 224) * 2;
    unsigned short* w4p   = (unsigned short*)ws; ws += 256 * (4 * 224) * 2;
    unsigned short* w5p   = (unsigned short*)ws; ws += 256 * (5 * 224) * 2;
    unsigned short* g1wt  = (unsigned short*)ws; ws += 256 * 128 * 2;   // N x K
    unsigned short* g2wt  = (unsigned short*)ws; ws += 768 * 256 * 2;   // N x K

    // zero max-pool target + segment-sum accumulators
    zero_kernel<<<256, 256, 0, stream>>>(out, 32 * 768);
    zero_kernel<<<2048, 256, 0, stream>>>(agg1, NN * 128);
    zero_kernel<<<2048, 256, 0, stream>>>(agg2, NN * 256);

    // one-time bf16 conversions / padded conv weights / transposed GCN weights
    convert_kernel<<<2048, 256, 0, stream>>>(emb, embbf, NN * 200);
    pad_convert_w_kernel<<<256, 256, 0, stream>>>(w3, w3p, 3);
    pad_convert_w_kernel<<<256, 256, 0, stream>>>(w4, w4p, 4);
    pad_convert_w_kernel<<<256, 256, 0, stream>>>(w5, w5p, 5);
    transpose_convert_kernel<<<128, 256, 0, stream>>>(g1w, g1wt, 128, 256);
    transpose_convert_kernel<<<256, 256, 0, stream>>>(g2w, g2wt, 256, 768);

    // TextCNN branch -> out rows [0, 32)
    conv_maxpool_kernel<<<dim3(128, 32, 3), 256, 0, stream>>>(
        input_seq, embbf, w3p, b3, w4p, b4, w5p, b5, out);

    // GCN layer 1
    scatter_add_kernel<<<4096, 256, 0, stream>>>(edge_src, edge_dst, features,
                                                 agg1, E, 128, 5);
    convert_kernel<<<2048, 256, 0, stream>>>(agg1, abf1, NN * 128);
    gemm_tdm_kernel<<<dim3((NN + 31) / 32, 256 / 64), 256, 0, stream>>>(
        abf1, g1wt, g1b, hbuf, NN, 128, 256, 256, 0, 1);

    // GCN layer 2 -> out rows [32, 50032)
    scatter_add_kernel<<<4096, 256, 0, stream>>>(edge_src, edge_dst, hbuf,
                                                 agg2, E, 256, 6);
    convert_kernel<<<2048, 256, 0, stream>>>(agg2, abf2, NN * 256);
    gemm_tdm_kernel<<<dim3((NN + 31) / 32, 768 / 64), 256, 0, stream>>>(
        abf2, g2wt, g2b, out, NN, 256, 768, 768, 32, 0);

    // log_softmax over all 50032 rows, in place
    logsoftmax_kernel<<<50032, 256, 0, stream>>>(out);
}
